// LinearAttention_2413771620387
// MI455X (gfx1250) — matrix operand order repaired and checked by reading, not compiled, not run on hardware
//
#include <hip/hip_runtime.h>
#include <hip/hip_bf16.h>

typedef __attribute__((ext_vector_type(16))) __bf16 v16bf;
typedef __attribute__((ext_vector_type(8)))  __bf16 v8bf;
typedef __attribute__((ext_vector_type(2)))  __bf16 v2bf;
typedef __attribute__((ext_vector_type(8)))  float  v8f;
typedef __attribute__((ext_vector_type(4)))  int    v4i;

#define CN   96
#define NTOK (64*64*32)   /* 131072 tokens per batch */
#define XS   104          /* [token][channel] LDS stride; f32 row = 416B (16B mult) */
#define TS   136          /* [channel][token] LDS stride (bf16), row = 272B (16B mult) */

#define AS1 __attribute__((address_space(1)))
#define AS3 __attribute__((address_space(3)))

#if __has_builtin(__builtin_amdgcn_global_load_async_to_lds_b32) && \
    __has_builtin(__builtin_amdgcn_global_store_async_from_lds_b128) && \
    __has_builtin(__builtin_amdgcn_s_wait_asynccnt)
#define USE_ASYNC 1
#else
#define USE_ASYNC 0
#endif

__device__ __forceinline__ v8f zero8() {
  v8f z;
#pragma unroll
  for (int i = 0; i < 8; ++i) z[i] = 0.f;
  return z;
}

// A operand (16x32 bf16) from bf16 LDS: lane m=lane&15 is row;
// per-lane K runs {h*8..h*8+7, 16+h*8..16+h*8+7}
__device__ __forceinline__ v16bf ldA(const __bf16* base, int rowStride, int rowBase,
                                     int kBase, int lane) {
  const int m = lane & 15, half = (lane >> 4) & 1;
  const __bf16* p = base + (rowBase + m) * rowStride + kBase + half * 8;
  v8bf lo = *(const v8bf*)(p);
  v8bf hi = *(const v8bf*)(p + 16);
  v16bf r;
#pragma unroll
  for (int i = 0; i < 8; ++i) { r[i] = lo[i]; r[i + 8] = hi[i]; }
  return r;
}

// A operand built from an fp32 LDS tile (converts on the fly)
__device__ __forceinline__ v16bf ldA_f32(const float* base, int rowStride, int rowBase,
                                         int kBase, int lane) {
  const int m = lane & 15, half = (lane >> 4) & 1;
  const float* p = base + (rowBase + m) * rowStride + kBase + half * 8;
  float4 a0 = *(const float4*)(p);
  float4 a1 = *(const float4*)(p + 4);
  float4 b0 = *(const float4*)(p + 16);
  float4 b1 = *(const float4*)(p + 20);
  v16bf r;
  r[0] = (__bf16)a0.x; r[1] = (__bf16)a0.y; r[2] = (__bf16)a0.z; r[3] = (__bf16)a0.w;
  r[4] = (__bf16)a1.x; r[5] = (__bf16)a1.y; r[6] = (__bf16)a1.z; r[7] = (__bf16)a1.w;
  r[8] = (__bf16)b0.x; r[9] = (__bf16)b0.y; r[10] = (__bf16)b0.z; r[11] = (__bf16)b0.w;
  r[12] = (__bf16)b1.x; r[13] = (__bf16)b1.y; r[14] = (__bf16)b1.z; r[15] = (__bf16)b1.w;
  return r;
}

// B operand (32x16 bf16): lane n=lane&15 is column; per-lane K runs {h*8..h*8+7, 16+h*8..16+h*8+7}
__device__ __forceinline__ v16bf ldB(const __bf16* base, int rowStride, int nBase,
                                     int kBase, int lane) {
  const int n = lane & 15, half = (lane >> 4) & 1;
  const __bf16* p = base + (nBase + n) * rowStride + kBase + half * 8;
  v8bf lo = *(const v8bf*)(p);
  v8bf hi = *(const v8bf*)(p + 16);
  v16bf r;
#pragma unroll
  for (int i = 0; i < 8; ++i) { r[i] = lo[i]; r[i + 8] = hi[i]; }
  return r;
}

__device__ __forceinline__ v8f wmma_bf16(v16bf a, v16bf b, v8f c) {
  return __builtin_amdgcn_wmma_f32_16x16x32_bf16(false, a, false, b, (short)0, c,
                                                 false, false);
}

// x tile -> LDS fp32, transposed [c][n]->[token][channel]; async when available
__device__ __forceinline__ void load_x_tile(const float* __restrict__ xb, int n0,
                                            int t0, int lane, float* xs) {
  const int t = lane & 15, ch = (lane >> 4) & 1;
#pragma unroll
  for (int i = 0; i < 12; ++i) {
    const int c = 4 * i + 2 * ch;
#if USE_ASYNC
    __builtin_amdgcn_global_load_async_to_lds_b32(
        (AS1 int*)&xb[(size_t)c * NTOK + n0 + t0 + t],
        (AS3 int*)&xs[(t0 + t) * XS + c], 0, 0);
    __builtin_amdgcn_global_load_async_to_lds_b32(
        (AS1 int*)&xb[(size_t)(c + 1) * NTOK + n0 + t0 + t],
        (AS3 int*)&xs[(t0 + t) * XS + c + 1], 0, 0);
#else
    float f0 = xb[(size_t)c * NTOK + n0 + t0 + t];
    float f1 = xb[(size_t)(c + 1) * NTOK + n0 + t0 + t];
    *(float2*)(&xs[(t0 + t) * XS + c]) = make_float2(f0, f1);
#endif
  }
#if USE_ASYNC
  __builtin_amdgcn_s_wait_asynccnt(0);
#endif
}

// ---------------------------------------------------------------------------
// Pass 1: K/V GEMMs (WMMA) + exp(K) ; context = exp(K)^T V via WMMA over tokens
// ---------------------------------------------------------------------------
__global__ void __launch_bounds__(256) lin_attn_pass1(
    const float* __restrict__ x, const float* __restrict__ Wk,
    const float* __restrict__ Wv, float* __restrict__ ctx_raw,
    float* __restrict__ s_sum) {
  extern __shared__ __align__(16) char smem[];
  __bf16* wk = (__bf16*)smem;              // 96*96 bf16
  __bf16* wv = wk + CN * CN;               // 96*96 bf16
  float*  xs = (float*)(wv + CN * CN);     // 128*XS f32 [token][channel]
  __bf16* ek = (__bf16*)(xs + 128 * XS);   // 96*TS bf16 [channel][token] exp(K)
  __bf16* vs = ek + CN * TS;               // 96*TS bf16 [channel][token] V

  const int tid = threadIdx.x, lane = tid & 31, wid = tid >> 5;

  for (int i = tid; i < CN * CN / 2; i += 256) {
    float2 a = ((const float2*)Wk)[i];
    float2 b = ((const float2*)Wv)[i];
    v2bf pa; pa[0] = (__bf16)a.x; pa[1] = (__bf16)a.y;
    v2bf pb; pb[0] = (__bf16)b.x; pb[1] = (__bf16)b.y;
    *(v2bf*)(wk + 2 * i) = pa;
    *(v2bf*)(wv + 2 * i) = pb;
  }

  const int wgPerB = gridDim.x >> 1;
  const int b  = blockIdx.x / wgPerB;
  const int wi = blockIdx.x % wgPerB;
  const float* xb = x + (size_t)b * CN * NTOK;

  v8f ctx[6];
#pragma unroll
  for (int j = 0; j < 6; ++j) ctx[j] = zero8();
  float sacc0 = 0.f, sacc1 = 0.f;

  const int t0 = wid * 16;

  for (int bi = wi; bi < NTOK / 128; bi += wgPerB) {
    const int n0 = bi * 128;
    load_x_tile(xb, n0, t0, lane, xs);
    if (bi + wgPerB < NTOK / 128)  // global_prefetch_b8 for next persistent tile
      __builtin_prefetch(&xb[(size_t)((lane >> 4) & 1) * NTOK +
                             (size_t)(bi + wgPerB) * 128 + t0 + (lane & 15)], 0, 0);
    __syncthreads();

    // --- K and V GEMMs (each wave: its 16 tokens x 96 outputs) ---
    v16bf A0 = ldA_f32(xs, XS, t0, 0, lane);
    v16bf A1 = ldA_f32(xs, XS, t0, 32, lane);
    v16bf A2 = ldA_f32(xs, XS, t0, 64, lane);
    const int cc = lane & 15, halfl = (lane >> 4) & 1;
#pragma unroll
    for (int j = 0; j < 6; ++j) {
      v8f acc = zero8();
      acc = wmma_bf16(A0, ldB(wk, CN, 16 * j, 0, lane), acc);
      acc = wmma_bf16(A1, ldB(wk, CN, 16 * j, 32, lane), acc);
      acc = wmma_bf16(A2, ldB(wk, CN, 16 * j, 64, lane), acc);
      v8bf pk;
#pragma unroll
      for (int r = 0; r < 8; ++r) pk[r] = (__bf16)__expf(acc[r]);
      *(v8bf*)(&ek[(16 * j + cc) * TS + t0 + 8 * halfl]) = pk;  // ds_store_b128
    }
#pragma unroll
    for (int j = 0; j < 6; ++j) {
      v8f acc = zero8();
      acc = wmma_bf16(A0, ldB(wv, CN, 16 * j, 0, lane), acc);
      acc = wmma_bf16(A1, ldB(wv, CN, 16 * j, 32, lane), acc);
      acc = wmma_bf16(A2, ldB(wv, CN, 16 * j, 64, lane), acc);
      v8bf pk;
#pragma unroll
      for (int r = 0; r < 8; ++r) pk[r] = (__bf16)acc[r];
      *(v8bf*)(&vs[(16 * j + cc) * TS + t0 + 8 * halfl]) = pk;
    }
    __syncthreads();

    // --- context accumulation: exp(K)^T * V, contraction over 128 tokens ---
    if (wid < 6) {
#pragma unroll
      for (int kk = 0; kk < 4; ++kk) {
        v16bf Ae = ldA(ek, TS, 16 * wid, 32 * kk, lane);
#pragma unroll
        for (int j = 0; j < 6; ++j)
          ctx[j] = wmma_bf16(Ae, ldB(vs, TS, 16 * j, 32 * kk, lane), ctx[j]);
      }
    } else {  // waves 6,7: column sums of exp(K) (softmax denominators)
      const int li = tid - 192;
      float a0 = 0.f;
#pragma unroll 4
      for (int q = 0; q < 16; ++q) {
        v8bf c8 = *(const v8bf*)(&ek[li * TS + 8 * q]);
#pragma unroll
        for (int e = 0; e < 8; ++e) a0 += (float)c8[e];
      }
      sacc0 += a0;
      if (li < 32) {
        float a1 = 0.f;
#pragma unroll 4
        for (int q = 0; q < 16; ++q) {
          v8bf c8 = *(const v8bf*)(&ek[(64 + li) * TS + 8 * q]);
#pragma unroll
          for (int e = 0; e < 8; ++e) a1 += (float)c8[e];
        }
        sacc1 += a1;
      }
    }
    __syncthreads();
  }

  // flush persistent accumulators (block-diagonal entries only)
  if (wid < 6) {
#pragma unroll
    for (int j = 0; j < 6; ++j) {
      const int c2 = 16 * j + (lane & 15);
      const int h2 = c2 / 12;
#pragma unroll
      for (int r = 0; r < 8; ++r) {
        const int c1 = 16 * wid + r + 8 * ((lane >> 4) & 1);
        const int h1 = c1 / 12;
        if (h1 == h2)
          atomicAdd(&ctx_raw[b * 1152 + (h1 * 12 + (c1 - 12 * h1)) * 12 + (c2 - 12 * h2)],
                    ctx[j][r]);
      }
    }
  } else {
    const int li = tid - 192;
    atomicAdd(&s_sum[b * CN + li], sacc0);
    if (li < 32) atomicAdd(&s_sum[b * CN + 64 + li], sacc1);
  }
}

// ---------------------------------------------------------------------------
// Pass 2: normalize context rows, emit zero-padded block-diagonal 96x96 bf16
// stored transposed [c2][c1] so it loads directly as a WMMA B operand.
// ---------------------------------------------------------------------------
__global__ void lin_attn_pass2(const float* __restrict__ ctx_raw,
                               const float* __restrict__ s_sum,
                               __bf16* __restrict__ cbdT) {
  int idx = blockIdx.x * 256 + threadIdx.x;
  if (idx >= 2 * CN * CN) return;
  int b = idx / (CN * CN);
  int r = idx % (CN * CN);
  int c2 = r / CN, c1 = r % CN;
  int h1 = c1 / 12, h2 = c2 / 12;
  float val = 0.f;
  if (h1 == h2) {
    float raw = ctx_raw[b * 1152 + (h1 * 12 + (c1 - 12 * h1)) * 12 + (c2 - 12 * h2)];
    val = raw / s_sum[b * CN + c1];
  }
  cbdT[(size_t)b * CN * CN + c2 * CN + c1] = (__bf16)val;
}

__global__ void lin_attn_init(float* __restrict__ ws_f) {
  int i = blockIdx.x * 256 + threadIdx.x;
  if (i < 2496) ws_f[i] = 0.f;
}

// ---------------------------------------------------------------------------
// Pass 3: Q GEMM -> softmax(12) -> attn = q @ C_bd (WMMA) -> proj GEMM -> out
// ---------------------------------------------------------------------------
__global__ void __launch_bounds__(256) lin_attn_pass3(
    const float* __restrict__ x, const float* __restrict__ Wq,
    const float* __restrict__ Wp, const __bf16* __restrict__ cbdT_g,
    float* __restrict__ out) {
  extern __shared__ __align__(16) char smem3[];
  __bf16* wq  = (__bf16*)smem3;            // 96*96 bf16
  __bf16* wp  = wq + CN * CN;              // 96*96 bf16
  __bf16* cb  = wp + CN * CN;              // 96*96 bf16  C_bd^T [c2][c1]
  float*  xs  = (float*)(cb + CN * CN);    // 128*XS f32
  __bf16* qs  = (__bf16*)(xs + 128 * XS);  // 128*XS bf16
  __bf16* as_ = qs + 128 * XS;             // 128*XS bf16
  float*  os  = (float*)xs;                // alias over xs only: 96*TS f32 fits

  const int tid = threadIdx.x, lane = tid & 31, wid = tid >> 5;
  const int blocksPerB = NTOK / 128;
  const int b  = blockIdx.x / blocksPerB;
  const int n0 = (blockIdx.x % blocksPerB) * 128;

  for (int i = tid; i < CN * CN / 2; i += 256) {
    float2 a = ((const float2*)Wq)[i];
    float2 c = ((const float2*)Wp)[i];
    v2bf pa; pa[0] = (__bf16)a.x; pa[1] = (__bf16)a.y;
    v2bf pc; pc[0] = (__bf16)c.x; pc[1] = (__bf16)c.y;
    *(v2bf*)(wq + 2 * i) = pa;
    *(v2bf*)(wp + 2 * i) = pc;
  }
  {
    const unsigned int* src = (const unsigned int*)(cbdT_g + (size_t)b * CN * CN);
    unsigned int* dst = (unsigned int*)cb;
    for (int i = tid; i < CN * CN / 2; i += 256) dst[i] = src[i];
  }
  const int t0 = wid * 16;
  load_x_tile(x + (size_t)b * CN * NTOK, n0, t0, lane, xs);
  __syncthreads();

  {  // Q GEMM
    v16bf A0 = ldA_f32(xs, XS, t0, 0, lane);
    v16bf A1 = ldA_f32(xs, XS, t0, 32, lane);
    v16bf A2 = ldA_f32(xs, XS, t0, 64, lane);
    const int cc = lane & 15, halfl = (lane >> 4) & 1;
#pragma unroll
    for (int j = 0; j < 6; ++j) {
      v8f acc = zero8();
      acc = wmma_bf16(A0, ldB(wq, CN, 16 * j, 0, lane), acc);
      acc = wmma_bf16(A1, ldB(wq, CN, 16 * j, 32, lane), acc);
      acc = wmma_bf16(A2, ldB(wq, CN, 16 * j, 64, lane), acc);
      const int c = 16 * j + cc, tb = t0 + 8 * halfl;
#pragma unroll
      for (int r = 0; r < 8; ++r) qs[(tb + r) * XS + c] = (__bf16)acc[r];
    }
  }
  __syncthreads();

  // softmax over head dim (12) per (token, head)
#pragma unroll
  for (int k = 0; k < 4; ++k) {
    const int rid = tid + 256 * k;  // 128 tokens * 8 heads = 1024 rows
    const int t = rid >> 3, h = rid & 7;
    __bf16* row = &qs[t * XS + 12 * h];
    float v[12];
    float m = -3.4e38f;
#pragma unroll
    for (int j = 0; j < 12; ++j) { v[j] = (float)row[j]; m = fmaxf(m, v[j]); }
    float ssum = 0.f;
#pragma unroll
    for (int j = 0; j < 12; ++j) { v[j] = __expf(v[j] - m); ssum += v[j]; }
    const float inv = 1.f / ssum;
#pragma unroll
    for (int j = 0; j < 12; ++j) row[j] = (__bf16)(v[j] * inv);
  }
  __syncthreads();

  {  // attn = q_soft @ C_bd  (block-diagonal 96x96 as a dense WMMA GEMM)
    v16bf A0 = ldA(qs, XS, t0, 0, lane);
    v16bf A1 = ldA(qs, XS, t0, 32, lane);
    v16bf A2 = ldA(qs, XS, t0, 64, lane);
    const int cc = lane & 15, halfl = (lane >> 4) & 1;
#pragma unroll
    for (int j = 0; j < 6; ++j) {
      v8f acc = zero8();
      acc = wmma_bf16(A0, ldB(cb, CN, 16 * j, 0, lane), acc);
      acc = wmma_bf16(A1, ldB(cb, CN, 16 * j, 32, lane), acc);
      acc = wmma_bf16(A2, ldB(cb, CN, 16 * j, 64, lane), acc);
      const int c = 16 * j + cc, tb = t0 + 8 * halfl;
#pragma unroll
      for (int r = 0; r < 8; ++r) as_[(tb + r) * XS + c] = (__bf16)acc[r];
    }
  }
  __syncthreads();

  {  // projection GEMM; stage results [o][t] for coalesced channel-major output
    v16bf A0 = ldA(as_, XS, t0, 0, lane);
    v16bf A1 = ldA(as_, XS, t0, 32, lane);
    v16bf A2 = ldA(as_, XS, t0, 64, lane);
    const int cc = lane & 15, halfl = (lane >> 4) & 1;
#pragma unroll
    for (int j = 0; j < 6; ++j) {
      v8f acc = zero8();
      acc = wmma_bf16(A0, ldB(wp, CN, 16 * j, 0, lane), acc);
      acc = wmma_bf16(A1, ldB(wp, CN, 16 * j, 32, lane), acc);
      acc = wmma_bf16(A2, ldB(wp, CN, 16 * j, 64, lane), acc);
      const int o = 16 * j + cc, tb = t0 + 8 * halfl;
      float4 lo = make_float4(acc[0], acc[1], acc[2], acc[3]);
      float4 hi = make_float4(acc[4], acc[5], acc[6], acc[7]);
      *(float4*)(&os[o * TS + tb]) = lo;
      *(float4*)(&os[o * TS + tb + 4]) = hi;
    }
  }
  __syncthreads();

#pragma unroll
  for (int k = 0; k < 12; ++k) {  // 96 rows * 32 float4 = 3072 / 256 threads
    const int idx = tid + 256 * k;
    const int o = idx >> 5;
    const int t = (idx & 31) * 4;
#if USE_ASYNC
    __builtin_amdgcn_global_store_async_from_lds_b128(
        (AS1 v4i*)&out[((size_t)(b * CN + o)) * NTOK + n0 + t],
        (AS3 v4i*)&os[o * TS + t], 0, 0);
#else
    float4 vv = *(const float4*)(&os[o * TS + t]);
    *(float4*)(&out[((size_t)(b * CN + o)) * NTOK + n0 + t]) = vv;
#endif
  }
#if USE_ASYNC
  __builtin_amdgcn_s_wait_asynccnt(0);
#endif
}

extern "C" void kernel_launch(void* const* d_in, const int* in_sizes, int n_in,
                              void* d_out, int out_size, void* d_ws, size_t ws_size,
                              hipStream_t stream) {
  (void)in_sizes; (void)n_in; (void)out_size; (void)ws_size;
  const float* x  = (const float*)d_in[0];
  const float* Wq = (const float*)d_in[1];
  const float* Wk = (const float*)d_in[2];
  const float* Wv = (const float*)d_in[3];
  const float* Wp = (const float*)d_in[4];
  float* out = (float*)d_out;

  float* ws_f    = (float*)d_ws;
  float* ctx_raw = ws_f;                       // [2][8*12*12] = 2304 f32
  float* s_sum   = ws_f + 2304;                // [2][96]      = 192 f32
  __bf16* cbdT   = (__bf16*)(ws_f + 2496);     // [2][96][96] bf16

  const size_t SM1 = (size_t)(2 * CN * CN + 2 * CN * TS) * sizeof(__bf16) +
                     (size_t)(128 * XS) * sizeof(float);
  const size_t SM3 = (size_t)(3 * CN * CN + 2 * 128 * XS) * sizeof(__bf16) +
                     (size_t)(128 * XS) * sizeof(float);
  (void)hipFuncSetAttribute((const void*)lin_attn_pass1,
                            hipFuncAttributeMaxDynamicSharedMemorySize, (int)SM1);
  (void)hipFuncSetAttribute((const void*)lin_attn_pass3,
                            hipFuncAttributeMaxDynamicSharedMemorySize, (int)SM3);

  lin_attn_init<<<10, 256, 0, stream>>>(ws_f);
  lin_attn_pass1<<<512, 256, SM1, stream>>>(x, Wk, Wv, ctx_raw, s_sum);
  lin_attn_pass2<<<(2 * CN * CN + 255) / 256, 256, 0, stream>>>(ctx_raw, s_sum, cbdT);
  lin_attn_pass3<<<2 * (NTOK / 128), 256, SM3, stream>>>(x, Wq, Wp, cbdT, out);
}
